// SelfAttention_51299089383807
// MI455X (gfx1250) — matrix operand order repaired and checked by reading, not compiled, not run on hardware
//
#include <hip/hip_runtime.h>

// ---------------------------------------------------------------------------
// Self-attention forward for MI455X (gfx1250), bf16 WMMA path.
//   B=4, S=2048, D = d_k = d_v = 1024
// Pass 1: Q/K/V projections fp32 -> bf16 (V written transposed)
// Pass 2: flash attention over bf16 Q/K/V, fp32 accumulation
// All GEMMs via v_wmma_f32_16x16x32_bf16; K-stream double-buffered in LDS.
// ---------------------------------------------------------------------------

typedef __attribute__((ext_vector_type(16))) __bf16 v16bf;
typedef __attribute__((ext_vector_type(8)))  float  v8f;

#define SEQ   2048
#define DIM   1024
#define ROWS  (4 * SEQ)   // B*S = 8192

// fp32 -> bf16 (round to nearest even), returned as raw bits
__device__ __forceinline__ unsigned short f2bf(float x) {
    unsigned u = __float_as_uint(x);
    u += 0x7fffu + ((u >> 16) & 1u);
    return (unsigned short)(u >> 16);
}

// Assemble a 16-element bf16 fragment from two 16-byte LDS chunks.
// off_* are in u16 units; both must be 8-element (16 B) aligned.
__device__ __forceinline__ v16bf load_frag(const unsigned short* row_base,
                                           int off_lo, int off_hi) {
    union { uint4 q[2]; v16bf v; } u;
    u.q[0] = *(const uint4*)(row_base + off_lo);
    u.q[1] = *(const uint4*)(row_base + off_hi);
    return u.v;
}

__device__ __forceinline__ v8f wmma_bf16(v16bf a, v16bf b, v8f c) {
    // (neg_a, A, neg_b, B, c_mod, C, reuse_a, reuse_b)
    return __builtin_amdgcn_wmma_f32_16x16x32_bf16(false, a, false, b,
                                                   (short)0, c, false, false);
}

// ---------------------------------------------------------------------------
// Projection GEMM:  C[8192,1024] = X[8192,1024] * W[1024,1024] + bias, -> bf16
// TR=false: out[m][n] (row-major);  TR=true: out[n][m] (transposed, for V).
// Workgroup: 256 threads = 8 waves; C tile 128x128; K-step 32, double-buffered.
// Wave w: rows 32*(w&3), cols 64*(w>>2) -> 2x4 WMMA tiles.
// ---------------------------------------------------------------------------
template <bool TR>
__global__ __launch_bounds__(256)
void proj_gemm_kernel(const float* __restrict__ X,
                      const float* __restrict__ W,
                      const float* __restrict__ bias,
                      unsigned short* __restrict__ out) {
    // double-buffered LDS tiles, contraction-minor, row stride 40 u16
    __shared__ unsigned short sA[2][128 * 40];  // [m][k]
    __shared__ unsigned short sB[2][128 * 40];  // [n][k]  (W transposed)

    const int t    = threadIdx.x;
    const int w    = t >> 5;
    const int lane = t & 31;
    const int h    = lane >> 4;
    const int l16  = lane & 15;

    const int m_base = blockIdx.x * 128;
    const int n_base = blockIdx.y * 128;
    const int tm = w & 3;
    const int tn = w >> 2;

    // per-thread load coordinates (1024 float4 chunks per tile)
    const int a_row = t >> 3, a_c4 = t & 7;          // +32 rows per i
    const int w_kr  = t >> 5, w_n4 = (t & 31) * 4;   // +8 k-rows per i

    v8f acc[2][4];
#pragma unroll
    for (int mi = 0; mi < 2; ++mi)
#pragma unroll
        for (int ni = 0; ni < 4; ++ni)
#pragma unroll
            for (int r = 0; r < 8; ++r) acc[mi][ni][r] = 0.0f;

    // ---- prologue: tile 0 -> buffer 0 ----
#pragma unroll
    for (int i = 0; i < 4; ++i) {
        float4 da = *(const float4*)(X + (size_t)(m_base + a_row + 32 * i) * DIM + a_c4 * 4);
        unsigned short* p = &sA[0][(a_row + 32 * i) * 40 + a_c4 * 4];
        p[0] = f2bf(da.x); p[1] = f2bf(da.y); p[2] = f2bf(da.z); p[3] = f2bf(da.w);
        float4 dw = *(const float4*)(W + (size_t)(w_kr + 8 * i) * DIM + n_base + w_n4);
        sB[0][(w_n4 + 0) * 40 + w_kr + 8 * i] = f2bf(dw.x);
        sB[0][(w_n4 + 1) * 40 + w_kr + 8 * i] = f2bf(dw.y);
        sB[0][(w_n4 + 2) * 40 + w_kr + 8 * i] = f2bf(dw.z);
        sB[0][(w_n4 + 3) * 40 + w_kr + 8 * i] = f2bf(dw.w);
    }
    __syncthreads();

    for (int c = 0; c < 32; ++c) {
        const int cur = c & 1, nxt = cur ^ 1;
        float4 ra[4], rw[4];
        if (c < 31) {            // stage next K-slab while WMMAs run
            int kk = (c + 1) * 32;
#pragma unroll
            for (int i = 0; i < 4; ++i) {
                ra[i] = *(const float4*)(X + (size_t)(m_base + a_row + 32 * i) * DIM + kk + a_c4 * 4);
                rw[i] = *(const float4*)(W + (size_t)(kk + w_kr + 8 * i) * DIM + n_base + w_n4);
            }
        }
        // ---- 8 WMMAs on current buffer ----
#pragma unroll
        for (int mi = 0; mi < 2; ++mi) {
            v16bf a = load_frag(&sA[cur][(32 * tm + 16 * mi + l16) * 40],
                                8 * h, 16 + 8 * h);
#pragma unroll
            for (int ni = 0; ni < 4; ++ni) {
                v16bf b = load_frag(&sB[cur][(64 * tn + 16 * ni + l16) * 40],
                                    8 * h, 16 + 8 * h);
                acc[mi][ni] = wmma_bf16(a, b, acc[mi][ni]);
            }
        }
        if (c < 31) {            // convert + commit staged slab
#pragma unroll
            for (int i = 0; i < 4; ++i) {
                unsigned short* p = &sA[nxt][(a_row + 32 * i) * 40 + a_c4 * 4];
                p[0] = f2bf(ra[i].x); p[1] = f2bf(ra[i].y);
                p[2] = f2bf(ra[i].z); p[3] = f2bf(ra[i].w);
                sB[nxt][(w_n4 + 0) * 40 + w_kr + 8 * i] = f2bf(rw[i].x);
                sB[nxt][(w_n4 + 1) * 40 + w_kr + 8 * i] = f2bf(rw[i].y);
                sB[nxt][(w_n4 + 2) * 40 + w_kr + 8 * i] = f2bf(rw[i].z);
                sB[nxt][(w_n4 + 3) * 40 + w_kr + 8 * i] = f2bf(rw[i].w);
            }
        }
        __syncthreads();
    }

    // ---- epilogue: + bias, fp32 -> bf16, store (optionally transposed) ----
#pragma unroll
    for (int mi = 0; mi < 2; ++mi) {
#pragma unroll
        for (int ni = 0; ni < 4; ++ni) {
            int n = n_base + 64 * tn + 16 * ni + l16;
            float bs = bias[n];
#pragma unroll
            for (int r = 0; r < 8; ++r) {
                int m = m_base + 32 * tm + 16 * mi + 8 * h + r;
                unsigned short v = f2bf(acc[mi][ni][r] + bs);
                if (TR) out[(size_t)n * ROWS + m] = v;
                else    out[(size_t)m * DIM  + n] = v;
            }
        }
    }
}

// ---------------------------------------------------------------------------
// Flash attention: workgroup = (32-query block, batch); 256 threads = 8 waves.
// Phase 1: wave (w&1, w>>1) computes one 16x16 tile of S[32,64], d_k streamed
//          in 128-wide chunks, double-buffered.
// Phase 2: online softmax, 8 threads/row + __shfl_xor reductions.
// Phase 3/4: each wave owns a 128-wide d_v slice; O[32,128] fp32 accumulators;
//            P*V WMMA over two 32-key halves, V already transposed in memory.
// ---------------------------------------------------------------------------
__global__ __launch_bounds__(256)
void flash_attn_kernel(const unsigned short* __restrict__ Qb,
                       const unsigned short* __restrict__ Kb,
                       const unsigned short* __restrict__ Vt,  // [v][B*S]
                       float* __restrict__ out) {
    __shared__ unsigned short sQ[32 * 1032];      // [m][k], stride 1032
    __shared__ unsigned short sK[2][64 * 136];    // [n][k-chunk], stride 136
    __shared__ unsigned short sVt[1024 * 40];     // [v][n-half], stride 40
    __shared__ float          sS[32 * 72];        // scaled scores
    __shared__ unsigned short sP[32 * 72];        // bf16 probabilities
    __shared__ float          sM[32], sL[32], sAlpha[32];

    const int b  = blockIdx.y;
    const int q0 = blockIdx.x * 32;
    const int t    = threadIdx.x;
    const int w    = t >> 5;
    const int lane = t & 31;
    const int h    = lane >> 4;
    const int l16  = lane & 15;
    const float scale = 0.03125f;  // 1/sqrt(1024)

    // K-chunk load coordinates: 1024 uint4 chunks -> 4 per thread
    const int k_row = t >> 2, k_c8 = (t & 3) * 4;  // rows +... see loop (i strides rows)
    // (use f = t + 256*i mapping below for clarity)

    // --- load Q tile (bf16): 32 x 1024 ---
#pragma unroll
    for (int i = 0; i < 16; ++i) {
        int f   = t + 256 * i;
        int row = f >> 7;
        int c8  = f & 127;
        uint4 d = *(const uint4*)(Qb + ((size_t)(b * SEQ + q0 + row)) * DIM + c8 * 8);
        *(uint4*)(sQ + row * 1032 + c8 * 8) = d;
    }
    if (t < 32) { sM[t] = -1e30f; sL[t] = 0.0f; }

    v8f acc[2][8];
#pragma unroll
    for (int mi = 0; mi < 2; ++mi)
#pragma unroll
        for (int ni = 0; ni < 8; ++ni)
#pragma unroll
            for (int r = 0; r < 8; ++r) acc[mi][ni][r] = 0.0f;

    const int tm = w & 1;
    const int tn = w >> 1;

    for (int j = 0; j < SEQ / 64; ++j) {
        const size_t kbase = (size_t)(b * SEQ + j * 64) * DIM;
        // ---- prologue: K chunk 0 -> sK[0] ----
#pragma unroll
        for (int i = 0; i < 4; ++i) {
            int f = t + 256 * i, row = f >> 4, c8 = f & 15;
            uint4 d = *(const uint4*)(Kb + kbase + (size_t)row * DIM + c8 * 8);
            *(uint4*)(&sK[0][row * 136 + c8 * 8]) = d;
        }
        __syncthreads();

        // ================= phase 1: S = Q K^T ============================
        v8f sacc;
#pragma unroll
        for (int r = 0; r < 8; ++r) sacc[r] = 0.0f;

        for (int c = 0; c < 8; ++c) {
            const int cur = c & 1, nxt = cur ^ 1;
            uint4 stg[4];
            if (c < 7) {
#pragma unroll
                for (int i = 0; i < 4; ++i) {
                    int f = t + 256 * i, row = f >> 4, c8 = f & 15;
                    stg[i] = *(const uint4*)(Kb + kbase + (size_t)row * DIM
                                             + (c + 1) * 128 + c8 * 8);
                }
            }
#pragma unroll
            for (int i = 0; i < 4; ++i) {
                int k0 = c * 128 + i * 32;
                v16bf a  = load_frag(sQ + (16 * tm + l16) * 1032,
                                     k0 + 8 * h, k0 + 16 + 8 * h);
                v16bf kb = load_frag(&sK[cur][(16 * tn + l16) * 136],
                                     i * 32 + 8 * h, i * 32 + 16 + 8 * h);
                sacc = wmma_bf16(a, kb, sacc);
            }
            if (c < 7) {
#pragma unroll
                for (int i = 0; i < 4; ++i) {
                    int f = t + 256 * i, row = f >> 4, c8 = f & 15;
                    *(uint4*)(&sK[nxt][row * 136 + c8 * 8]) = stg[i];
                }
            }
            __syncthreads();
        }
        // write scaled score tile (C layout: vgpr r -> row r+8h, col l16)
#pragma unroll
        for (int r = 0; r < 8; ++r)
            sS[(16 * tm + 8 * h + r) * 72 + 16 * tn + l16] = sacc[r] * scale;
        __syncthreads();

        // ================= phase 2: online softmax (8 thr/row) ===========
        {
            int row = t >> 3;          // 0..31
            int seg = t & 7;           // 8 cols each; same-wave lane groups
            float mold = sM[row];
            float sv[8], pmax = -1e30f;
#pragma unroll
            for (int e = 0; e < 8; ++e) {
                sv[e] = sS[row * 72 + seg * 8 + e];
                pmax  = fmaxf(pmax, sv[e]);
            }
            pmax = fmaxf(pmax, __shfl_xor(pmax, 1, 32));
            pmax = fmaxf(pmax, __shfl_xor(pmax, 2, 32));
            pmax = fmaxf(pmax, __shfl_xor(pmax, 4, 32));
            float mnew  = fmaxf(mold, pmax);
            float alpha = __expf(mold - mnew);
            float psum  = 0.0f;
#pragma unroll
            for (int e = 0; e < 8; ++e) {
                float p = __expf(sv[e] - mnew);
                psum += p;
                sP[row * 72 + seg * 8 + e] = f2bf(p);
            }
            psum += __shfl_xor(psum, 1, 32);
            psum += __shfl_xor(psum, 2, 32);
            psum += __shfl_xor(psum, 4, 32);
            if (seg == 0) {
                sL[row]     = alpha * sL[row] + psum;
                sM[row]     = mnew;
                sAlpha[row] = alpha;
            }
        }
        __syncthreads();

        // prefetch next key block into L2/WGP$ while we rescale
        if (j + 1 < SEQ / 64) {
            const unsigned short* nk = Kb + (size_t)(b * SEQ + (j + 1) * 64 + (t >> 2)) * DIM
                                          + (size_t)(t & 3) * 32;
            __builtin_prefetch(nk, 0, 0);
        }

        // ================= phase 3: rescale O accumulators ===============
#pragma unroll
        for (int mi = 0; mi < 2; ++mi) {
            float al[8];
#pragma unroll
            for (int r = 0; r < 8; ++r) al[r] = sAlpha[16 * mi + 8 * h + r];
#pragma unroll
            for (int ni = 0; ni < 8; ++ni)
#pragma unroll
                for (int r = 0; r < 8; ++r) acc[mi][ni][r] *= al[r];
        }

        // ================= phase 4: O += P * V (two 32-key halves) =======
        for (int nh = 0; nh < 2; ++nh) {
            __syncthreads();  // sVt safe to overwrite
            const size_t m0 = (size_t)(b * SEQ + j * 64 + nh * 32);
            // V^T tile: 1024 v-rows x 32 keys, contiguous copy (no transpose)
#pragma unroll
            for (int i = 0; i < 16; ++i) {
                int f  = t + 256 * i;    // 0..4095 chunks (4 per v-row)
                int v  = f >> 2;
                int c4 = f & 3;
                uint4 d = *(const uint4*)(Vt + (size_t)v * ROWS + m0 + c4 * 8);
                *(uint4*)(sVt + v * 40 + c4 * 8) = d;
            }
            __syncthreads();
#pragma unroll
            for (int mi = 0; mi < 2; ++mi) {
                v16bf a = load_frag(sP + (16 * mi + l16) * 72,
                                    nh * 32 + 8 * h, nh * 32 + 16 + 8 * h);
#pragma unroll
                for (int ni = 0; ni < 8; ++ni) {
                    v16bf vb = load_frag(sVt + (w * 128 + 16 * ni + l16) * 40,
                                         8 * h, 16 + 8 * h);
                    acc[mi][ni] = wmma_bf16(a, vb, acc[mi][ni]);
                }
            }
        }
    }

    // ================= epilogue: O / l, fp32 store =======================
#pragma unroll
    for (int mi = 0; mi < 2; ++mi) {
        float linv[8];
#pragma unroll
        for (int r = 0; r < 8; ++r) linv[r] = 1.0f / sL[16 * mi + 8 * h + r];
#pragma unroll
        for (int ni = 0; ni < 8; ++ni) {
#pragma unroll
            for (int r = 0; r < 8; ++r) {
                int m = 16 * mi + 8 * h + r;
                int v = w * 128 + 16 * ni + l16;
                out[((size_t)(b * SEQ + q0 + m)) * DIM + v] = acc[mi][ni][r] * linv[r];
            }
        }
    }
}

// ---------------------------------------------------------------------------
extern "C" void kernel_launch(void* const* d_in, const int* in_sizes, int n_in,
                              void* d_out, int out_size, void* d_ws, size_t ws_size,
                              hipStream_t stream) {
    const float* q_in = (const float*)d_in[0];
    const float* k_in = (const float*)d_in[1];
    const float* v_in = (const float*)d_in[2];
    const float* Wq   = (const float*)d_in[3];
    const float* bq   = (const float*)d_in[4];
    const float* Wk   = (const float*)d_in[5];
    const float* bk   = (const float*)d_in[6];
    const float* Wv   = (const float*)d_in[7];
    const float* bv   = (const float*)d_in[8];

    // workspace: bf16 Q[m][k], K[m][k], V^T[v][m]; 16 MB each = 48 MB
    unsigned short* Qb  = (unsigned short*)d_ws;
    unsigned short* Kb  = Qb + (size_t)ROWS * DIM;
    unsigned short* VbT = Kb + (size_t)ROWS * DIM;

    dim3 gp(ROWS / 128, DIM / 128);        // (64, 8)
    proj_gemm_kernel<false><<<gp, 256, 0, stream>>>(q_in, Wq, bq, Qb);
    proj_gemm_kernel<false><<<gp, 256, 0, stream>>>(k_in, Wk, bk, Kb);
    proj_gemm_kernel<true ><<<gp, 256, 0, stream>>>(v_in, Wv, bv, VbT);

    dim3 ga(SEQ / 32, 4);                  // (64, 4)
    flash_attn_kernel<<<ga, 256, 0, stream>>>(Qb, Kb, VbT, (float*)d_out);
}
